// ShootInv_37091337568587
// MI455X (gfx1250) — matrix operand order repaired and checked by reading, not compiled, not run on hardware
//
#include <hip/hip_runtime.h>
#include <math.h>

// ---------------------------------------------------------------------------
// LDDMM geodesic shooting (ShootInv) for MI455X / gfx1250.
// Spectral operator = batched 192-point DFT matmuls on the f16 WMMA pipe
// (f32 accumulate): v_wmma_f32_16x16x32_f16. B tiles staged in LDS in
// fragment-swizzled layout (read back once as ds_load_b128 into registers);
// A fragments come straight from global as 16B runs (global_load_b128),
// staged in one clause per k-tile so all 24 WMMAs issue back-to-back.
// ---------------------------------------------------------------------------

#define N1   192
#define LSTR (192 * 192)            // 36864 lines per channel
#define VOL  (192 * 192 * 192)      // 7,077,888 voxels
#define STEPS 8
#define DT    (1.0f / 8.0f)
#define ALPHA 1e-3f
#define LAM   1.0f

typedef __attribute__((ext_vector_type(16))) _Float16 v16h;
typedef __attribute__((ext_vector_type(8)))  _Float16 v8h;
typedef __attribute__((ext_vector_type(8)))  float    v8f;

// ---------------------------------------------------------------------------
// Table generation: DFT cos/sin matrices (f16) and 1-D Laplacian eigenvalues.
// ---------------------------------------------------------------------------
__global__ void gen_tables_kernel(_Float16* __restrict__ Ct,
                                  _Float16* __restrict__ St,
                                  float* __restrict__ etab) {
    int i = blockIdx.x * blockDim.x + threadIdx.x;
    if (i < N1) {
        float ang = 6.28318530717958647692f * (float)i / (float)N1;
        etab[i] = 2.0f - 2.0f * cosf(ang);
    }
    if (i < N1 * N1) {
        int k = i / N1, n = i % N1;
        int p = (k * n) % N1;                      // exact periodic reduction
        float ang = 6.28318530717958647692f * (float)p / (float)N1;
        Ct[i] = (_Float16)cosf(ang);
        St[i] = (_Float16)sinf(ang);
    }
}

__global__ void zero_kernel(float* __restrict__ p, size_t n) {
    size_t i = (size_t)blockIdx.x * blockDim.x + threadIdx.x;
    if (i < n) p[i] = 0.0f;
}

// ---------------------------------------------------------------------------
// One DFT pass along the innermost axis, expressed as WMMA GEMM.
//   out[k*36864 + l] = sum_n M[k,n] * in[l*192 + n]      (per channel)
// The write pattern rotates the volume axes; 3 passes cycle back to identity.
// mode: 0 = forward, real input      (out_re =  C f,          out_im = -S f)
//       1 = forward, complex         (out_re =  C re + S im,  out_im = -S re + C im)
//       2 = inverse, complex         (out_re =  C re - S im,  out_im =  S re + C im)
//       3 = inverse, real output     (out_re =  C re - S im;  no im store)
//
// One wave per block. Block covers 16 lines x 96 output rows (6 k-tiles,
// k-split 2 via blockIdx.y). B (16 lines x full K=192, re+im) is staged once
// in LDS, pre-swizzled to the wave32 B-fragment layout, then held in
// registers for the whole k-tile loop.
// ---------------------------------------------------------------------------
__global__ __launch_bounds__(32)
void dft_pass_kernel(const _Float16* __restrict__ Ct,
                     const _Float16* __restrict__ St,
                     const float* __restrict__ in_re,
                     const float* __restrict__ in_im,
                     float* __restrict__ out_re,
                     float* __restrict__ out_im,
                     int mode) {
    // [chunk 0..5][dlane 0..31][e 0..15], fragment-swizzled
    __shared__ __attribute__((aligned(32))) _Float16 LBre[6 * 512];
    __shared__ __attribute__((aligned(32))) _Float16 LBim[6 * 512];

    const int lane = threadIdx.x;
    const int l0 = blockIdx.x * 16;            // line tile
    const int kt0 = blockIdx.y * 6;            // k-tile range [kt0, kt0+6)
    const int ch = blockIdx.z;

    const float c_si = (mode == 0) ? 0.0f : ((mode == 1) ? 1.0f : -1.0f);
    const float c_sr = (mode <= 1) ? -1.0f : 1.0f;
    const float c_ci = (mode == 0) ? 0.0f : 1.0f;
    const bool  wim  = (mode != 3);

    const float* ire = in_re + (size_t)ch * VOL;
    const float* iim = in_im + (size_t)ch * VOL;

    // --- stage B into fragment layout ------------------------------------
    // B fragment, k order used here (contiguous, as A): lanes 0-15 -> N=lane, K=e (0..15);
    //                                lanes 16-31 -> N=lane-16, K=16+e.
    // Task (dlane, chunk): read 16 consecutive floats of line (dlane&15)
    // at n = chunk*32 + (dlane&16), write one 32B contiguous LDS run.
    #pragma unroll
    for (int it = 0; it < 6; ++it) {
        int task = it * 32 + lane;             // 0..191
        int dlane = task & 31;
        int c = task >> 5;                     // chunk 0..5
        int l = dlane & 15;
        int nbase = c * 32 + ((dlane & 16) ? 16 : 0);
        const float4* pr = (const float4*)&ire[(size_t)(l0 + l) * N1 + nbase];
        const float4* pi = (const float4*)&iim[(size_t)(l0 + l) * N1 + nbase];
        v16h hr, hi;
        #pragma unroll
        for (int q = 0; q < 4; ++q) {
            float4 fr = pr[q], fi = pi[q];
            hr[4 * q + 0] = (_Float16)fr.x; hi[4 * q + 0] = (_Float16)fi.x;
            hr[4 * q + 1] = (_Float16)fr.y; hi[4 * q + 1] = (_Float16)fi.y;
            hr[4 * q + 2] = (_Float16)fr.z; hi[4 * q + 2] = (_Float16)fi.z;
            hr[4 * q + 3] = (_Float16)fr.w; hi[4 * q + 3] = (_Float16)fi.w;
        }
        *(v16h*)&LBre[c * 512 + dlane * 16] = hr;
        *(v16h*)&LBim[c * 512 + dlane * 16] = hi;
    }
    __syncthreads();

    // --- pull B fragments into registers once (2 ds_load_b128 each) ------
    v16h bRf[6], bIf[6];
    #pragma unroll
    for (int cc = 0; cc < 6; ++cc) {
        bRf[cc] = *(const v16h*)&LBre[cc * 512 + lane * 16];
        bIf[cc] = *(const v16h*)&LBim[cc * 512 + lane * 16];
    }

    // --- compute: 6 k-tiles x (stage 12 A fragments, then 24 WMMA) --------
    // A fragment, k order used here (contiguous, as B): lanes 0-15 -> M=lane, K in {0..15};
    //                         lanes 16-31 -> M=lane-16, K in {16..31}.
    // Per lane: one contiguous 16-half run in the row-major table.
    const int am = lane & 15;
    const int ab = (lane < 16) ? 0 : 16;
    const int nn = lane & 15;
    const int mb = (lane < 16) ? 0 : 8;
    float* ore = out_re + (size_t)ch * VOL;
    float* oim = out_im + (size_t)ch * VOL;

    for (int kt = kt0; kt < kt0 + 6; ++kt) {
        int k0 = kt * 16;
        const _Float16* arow_c = &Ct[(size_t)(k0 + am) * N1 + ab];
        const _Float16* arow_s = &St[(size_t)(k0 + am) * N1 + ab];
        // prefetch next k-tile's table rows (global_prefetch_b8)
        __builtin_prefetch(arow_c + 16 * N1, 0, 1);

        // phase 1: stage all A fragments for this k-tile (one load clause)
        v16h aCf[6], aSf[6];
        #pragma unroll
        for (int cc = 0; cc < 6; ++cc) {
            v8h c_lo = *(const v8h*)(arow_c + cc * 32);
            v8h c_hi = *(const v8h*)(arow_c + cc * 32 + 8);
            v8h s_lo = *(const v8h*)(arow_s + cc * 32);
            v8h s_hi = *(const v8h*)(arow_s + cc * 32 + 8);
            aCf[cc] = __builtin_shufflevector(c_lo, c_hi, 0, 1, 2, 3, 4, 5, 6, 7,
                                              8, 9, 10, 11, 12, 13, 14, 15);
            aSf[cc] = __builtin_shufflevector(s_lo, s_hi, 0, 1, 2, 3, 4, 5, 6, 7,
                                              8, 9, 10, 11, 12, 13, 14, 15);
        }

        // phase 2: 24 back-to-back WMMAs on 4 independent accumulator chains
        v8f aCR = {}, aSI = {}, aSR = {}, aCI = {};
        #pragma unroll
        for (int cc = 0; cc < 6; ++cc) {
            aCR = __builtin_amdgcn_wmma_f32_16x16x32_f16(false, aCf[cc], false, bRf[cc],
                                                         (short)0, aCR, false, false);
            aSI = __builtin_amdgcn_wmma_f32_16x16x32_f16(false, aSf[cc], false, bIf[cc],
                                                         (short)0, aSI, false, false);
            aSR = __builtin_amdgcn_wmma_f32_16x16x32_f16(false, aSf[cc], false, bRf[cc],
                                                         (short)0, aSR, false, false);
            aCI = __builtin_amdgcn_wmma_f32_16x16x32_f16(false, aCf[cc], false, bIf[cc],
                                                         (short)0, aCI, false, false);
        }

        // C/D layout: lanes 0-15 N=lane rows M=0..7; lanes 16-31 N=lane-16,
        // rows M=8..15; VGPR r -> M = mb + r.
        #pragma unroll
        for (int r = 0; r < 8; ++r) {
            int krow = k0 + mb + r;
            size_t o = (size_t)krow * LSTR + (size_t)(l0 + nn);
            ore[o] = aCR[r] + c_si * aSI[r];
            if (wim) oim[o] = c_sr * aSR[r] + c_ci * aCI[r];
        }
    }
}

// ---------------------------------------------------------------------------
// Pointwise spectral scale: (alpha + lam*(e_x+e_y+e_z))^{+-2} / N^3.
// powmode=1 -> L_hat (sharpen, m0 = L v); powmode=0 -> K_hat (smooth, v = K m).
// ---------------------------------------------------------------------------
__global__ void scale_kernel(float* __restrict__ re, float* __restrict__ im,
                             const float* __restrict__ etab, int powmode) {
    size_t idx = (size_t)blockIdx.x * blockDim.x + threadIdx.x;
    if (idx >= (size_t)3 * VOL) return;
    size_t iv = idx % (size_t)VOL;
    int i2 = (int)(iv % N1);
    int i1 = (int)((iv / N1) % N1);
    int i0 = (int)(iv / (N1 * N1));
    float L = ALPHA + LAM * (etab[i0] + etab[i1] + etab[i2]);
    float L2 = L * L;
    float f = powmode ? L2 : (1.0f / L2);
    f *= 1.0f / ((float)N1 * (float)N1 * (float)N1);   // inverse-FFT norm
    re[idx] *= f;
    im[idx] *= f;
}

// ---------------------------------------------------------------------------
// u_new = trilinear_wrap(u_old, grid - dt*vel) - dt*vel
// ---------------------------------------------------------------------------
__global__ void pull_kernel(const float* __restrict__ u,
                            const float* __restrict__ vel,
                            float* __restrict__ uo) {
    size_t idx = (size_t)blockIdx.x * blockDim.x + threadIdx.x;
    if (idx >= (size_t)VOL) return;
    int z = (int)(idx % N1);
    int y = (int)((idx / N1) % N1);
    int x = (int)(idx / (N1 * N1));

    float vx = vel[0 * (size_t)VOL + idx];
    float vy = vel[1 * (size_t)VOL + idx];
    float vz = vel[2 * (size_t)VOL + idx];

    float cx = (float)x - DT * vx;
    float cy = (float)y - DT * vy;
    float cz = (float)z - DT * vz;

    int ix = (int)floorf(cx); float fx = cx - (float)ix;
    int iy = (int)floorf(cy); float fy = cy - (float)iy;
    int iz = (int)floorf(cz); float fz = cz - (float)iz;
    ix %= N1; if (ix < 0) ix += N1;
    iy %= N1; if (iy < 0) iy += N1;
    iz %= N1; if (iz < 0) iz += N1;
    int ixp = (ix + 1 == N1) ? 0 : ix + 1;
    int iyp = (iy + 1 == N1) ? 0 : iy + 1;
    int izp = (iz + 1 == N1) ? 0 : iz + 1;

    float w000 = (1 - fx) * (1 - fy) * (1 - fz);
    float w001 = (1 - fx) * (1 - fy) * fz;
    float w010 = (1 - fx) * fy * (1 - fz);
    float w011 = (1 - fx) * fy * fz;
    float w100 = fx * (1 - fy) * (1 - fz);
    float w101 = fx * (1 - fy) * fz;
    float w110 = fx * fy * (1 - fz);
    float w111 = fx * fy * fz;

    #pragma unroll
    for (int d = 0; d < 3; ++d) {
        const float* ud = u + (size_t)d * VOL;
        #define AT(X, Y, Z) ud[((size_t)(X) * N1 + (Y)) * N1 + (Z)]
        float s = w000 * AT(ix, iy, iz)  + w001 * AT(ix, iy, izp)
                + w010 * AT(ix, iyp, iz) + w011 * AT(ix, iyp, izp)
                + w100 * AT(ixp, iy, iz) + w101 * AT(ixp, iy, izp)
                + w110 * AT(ixp, iyp, iz)+ w111 * AT(ixp, iyp, izp);
        #undef AT
        uo[(size_t)d * VOL + idx] = s - DT * vel[(size_t)d * VOL + idx];
    }
}

// ---------------------------------------------------------------------------
// EPDiff Euler step: mo = m - dt*( (Dv)^T m + div(m x v) ), periodic central diff.
// ---------------------------------------------------------------------------
__global__ void epdiff_kernel(const float* __restrict__ m,
                              const float* __restrict__ vel,
                              float* __restrict__ mo) {
    size_t idx = (size_t)blockIdx.x * blockDim.x + threadIdx.x;
    if (idx >= (size_t)VOL) return;
    int z = (int)(idx % N1);
    int y = (int)((idx / N1) % N1);
    int x = (int)(idx / (N1 * N1));
    int xp = (x + 1 == N1) ? 0 : x + 1, xm = (x == 0) ? N1 - 1 : x - 1;
    int yp = (y + 1 == N1) ? 0 : y + 1, ym = (y == 0) ? N1 - 1 : y - 1;
    int zp = (z + 1 == N1) ? 0 : z + 1, zm = (z == 0) ? N1 - 1 : z - 1;

    #define IDX3(X, Y, Z) (((size_t)(X) * N1 + (Y)) * N1 + (Z))
    size_t ic = IDX3(x, y, z);
    size_t nb[3][2] = { { IDX3(xp, y, z), IDX3(xm, y, z) },
                        { IDX3(x, yp, z), IDX3(x, ym, z) },
                        { IDX3(x, y, zp), IDX3(x, y, zm) } };
    #undef IDX3

    float ml[3];
    #pragma unroll
    for (int j = 0; j < 3; ++j) ml[j] = m[(size_t)j * VOL + ic];

    #pragma unroll
    for (int d = 0; d < 3; ++d) {
        float t1 = 0.0f, t2 = 0.0f;
        #pragma unroll
        for (int j = 0; j < 3; ++j) {
            // t1 += grad_d(vel_j) * m_j
            t1 += 0.5f * (vel[(size_t)j * VOL + nb[d][0]] -
                          vel[(size_t)j * VOL + nb[d][1]]) * ml[j];
            // t2 += grad_j(m_d * vel_j)
            t2 += 0.5f * (m[(size_t)d * VOL + nb[j][0]] * vel[(size_t)j * VOL + nb[j][0]] -
                          m[(size_t)d * VOL + nb[j][1]] * vel[(size_t)j * VOL + nb[j][1]]);
        }
        mo[(size_t)d * VOL + ic] = ml[d] - DT * (t1 + t2);
    }
}

// ---------------------------------------------------------------------------
// Host-side operator composition
// ---------------------------------------------------------------------------
static void apply_operator(const float* in, float* out, int powmode,
                           const _Float16* Ct, const _Float16* St, const float* etab,
                           float* t_re, float* t_im, float* s_re, float* s_im,
                           hipStream_t stream) {
    dim3 g(LSTR / 16, 2, 3);      // 2304 line-tiles x 2 k-splits x 3 channels
    dim3 b(32, 1, 1);
    // forward: 3 axis passes (layout rotates each pass; 3 passes = identity)
    dft_pass_kernel<<<g, b, 0, stream>>>(Ct, St, in,   in,   t_re, t_im, 0);
    dft_pass_kernel<<<g, b, 0, stream>>>(Ct, St, t_re, t_im, s_re, s_im, 1);
    dft_pass_kernel<<<g, b, 0, stream>>>(Ct, St, s_re, s_im, t_re, t_im, 1);
    // spectral scale
    size_t n3 = (size_t)3 * VOL;
    scale_kernel<<<(unsigned)((n3 + 255) / 256), 256, 0, stream>>>(t_re, t_im, etab, powmode);
    // inverse: 3 axis passes, last emits real part only
    dft_pass_kernel<<<g, b, 0, stream>>>(Ct, St, t_re, t_im, s_re, s_im, 2);
    dft_pass_kernel<<<g, b, 0, stream>>>(Ct, St, s_re, s_im, t_re, t_im, 2);
    dft_pass_kernel<<<g, b, 0, stream>>>(Ct, St, t_re, t_im, out,  s_im, 3);
}

extern "C" void kernel_launch(void* const* d_in, const int* in_sizes, int n_in,
                              void* d_out, int out_size, void* d_ws, size_t ws_size,
                              hipStream_t stream) {
    (void)in_sizes; (void)n_in; (void)out_size; (void)ws_size;
    const float* v = (const float*)d_in[0];
    float* out = (float*)d_out;
    char* ws = (char*)d_ws;

    // workspace layout
    _Float16* Ct   = (_Float16*)(ws);
    _Float16* St   = (_Float16*)(ws + 128 * 1024);
    float*    etab = (float*)(ws + 256 * 1024);
    float*    vols = (float*)(ws + (1 << 20));
    const size_t V3 = (size_t)3 * VOL;
    float* m_a  = vols;
    float* m_b  = m_a + V3;
    float* u_a  = m_b + V3;
    float* u_b  = u_a + V3;
    float* vel  = u_b + V3;
    float* t_re = vel + V3;
    float* t_im = t_re + V3;
    float* s_re = t_im + V3;
    float* s_im = s_re + V3;

    // tables + init
    gen_tables_kernel<<<(N1 * N1 + 255) / 256, 256, 0, stream>>>(Ct, St, etab);
    zero_kernel<<<(unsigned)((V3 + 255) / 256), 256, 0, stream>>>(u_a, V3);

    // m0 = L v  (sharpen)
    apply_operator(v, m_a, /*powmode=*/1, Ct, St, etab, t_re, t_im, s_re, s_im, stream);

    float* m_cur = m_a; float* m_nxt = m_b;
    float* u_cur = u_a; float* u_nxt = u_b;
    const unsigned pb = (unsigned)(((size_t)VOL + 255) / 256);

    for (int step = 0; step < STEPS; ++step) {
        // vel = K m  (smooth)
        apply_operator(m_cur, vel, /*powmode=*/0, Ct, St, etab,
                       t_re, t_im, s_re, s_im, stream);
        // u <- pull(u, id - dt v) - dt v ; final step writes straight to d_out
        float* u_dst = (step == STEPS - 1) ? out : u_nxt;
        pull_kernel<<<pb, 256, 0, stream>>>(u_cur, vel, u_dst);
        // m <- m - dt * EPDiff(m, vel)  (skip on last step; result unused)
        if (step < STEPS - 1) {
            epdiff_kernel<<<pb, 256, 0, stream>>>(m_cur, vel, m_nxt);
            float* tm = m_cur; m_cur = m_nxt; m_nxt = tm;
        }
        float* tu = u_cur; u_cur = u_nxt; u_nxt = tu;
    }
}
